// WhisperAttention_90984587198783
// MI455X (gfx1250) — matrix operand order repaired and checked by reading, not compiled, not run on hardware
//
#include <hip/hip_runtime.h>
#include <stdint.h>

typedef __attribute__((ext_vector_type(16))) __bf16  v16bf;
typedef __attribute__((ext_vector_type(8)))  __bf16  v8bf;
typedef __attribute__((ext_vector_type(8)))  float   v8f;
typedef __attribute__((ext_vector_type(4)))  float   v4f;

static __device__ __forceinline__ __bf16 f32_to_bf16(float f) {
  uint32_t u = __builtin_bit_cast(uint32_t, f);
  u += 0x7FFFu + ((u >> 16) & 1u);            // round-to-nearest-even
  uint16_t h = (uint16_t)(u >> 16);
  return __builtin_bit_cast(__bf16, h);
}

// Workgroup-relative LDS byte offset of a generic pointer into a __shared__ array.
static __device__ __forceinline__ unsigned lds_off_of(const void* p) {
  return (unsigned)(unsigned long long)(__attribute__((address_space(3))) const char*)p;
}

// Async 16B/lane copy global -> LDS (CDNA5, tracked by ASYNCcnt).
static __device__ __forceinline__ void async_ld16(unsigned lds_off, const void* g) {
  asm volatile("global_load_async_to_lds_b128 %0, %1, off"
               :: "v"(lds_off), "v"((unsigned long long)(uintptr_t)g)
               : "memory");
}

static __device__ __forceinline__ void wait_async0() {
#if __has_builtin(__builtin_amdgcn_s_wait_asynccnt)
  __builtin_amdgcn_s_wait_asynccnt(0);
#else
  asm volatile("s_wait_asynccnt 0x0" ::: "memory");
#endif
}

#define BM 128
#define BN 128
#define BK 32
#define LPAD 8
#define TPB 256

// C[M,N] = scale * (A[MxK] @ Bt[NxK]^T) + bias[n]
// OUT_MODE 0: f32 C (ldc=N);  1: bf16 C (ldc=N);
// OUT_MODE 2: bf16 C written transposed per batch: C[b][n][s], b=m/S, s=m%S
template<int A_F32, int B_F32, int OUT_MODE>
__global__ __launch_bounds__(TPB)
void gemm_bt_kernel(const void* __restrict__ Ap, const void* __restrict__ Bp,
                    void* __restrict__ Cp, const float* __restrict__ bias,
                    float scale, int M, int N, int K, int S)
{
  __shared__ __bf16 As[2][BM][BK + LPAD];
  __shared__ __bf16 Bs[2][BN][BK + LPAD];

  const int tid  = threadIdx.x;
  const int lane = tid & 31;
  const int w    = tid >> 5;       // 8 waves
  const int wm   = w & 1;          // 2 wave-rows  (64 M rows each)
  const int wn   = w >> 1;         // 4 wave-cols  (32 N cols each)
  const int half = lane >> 4;      // 0/1
  const int lr   = lane & 15;

  const int m0 = blockIdx.y * BM;
  const int n0 = blockIdx.x * BN;

  const int srow = tid >> 1;       // 0..127 staging row
  const int sseg = tid & 1;        // 16-element column segment

  v8f acc[4][2] = {};
  v4f regA[4];                     // f32-source prefetch registers
  v4f regB[4];

  // ---------- staging helpers ----------
  auto stageA_issue = [&](int buf, int k0) {
    if (A_F32) {
      const float* src = (const float*)Ap + (size_t)(m0 + srow) * K + k0 + sseg * 16;
#pragma unroll
      for (int j = 0; j < 4; ++j) regA[j] = *(const v4f*)(src + 4 * j);
    } else {
      const __bf16* src = (const __bf16*)Ap + (size_t)(m0 + srow) * K + k0 + sseg * 16;
      async_ld16(lds_off_of(&As[buf][srow][sseg * 16]),     src);
      async_ld16(lds_off_of(&As[buf][srow][sseg * 16 + 8]), src + 8);
    }
  };
  auto stageB_issue = [&](int buf, int k0) {
    if (B_F32) {
      const float* src = (const float*)Bp + (size_t)(n0 + srow) * K + k0 + sseg * 16;
#pragma unroll
      for (int j = 0; j < 4; ++j) regB[j] = *(const v4f*)(src + 4 * j);
    } else {
      const __bf16* src = (const __bf16*)Bp + (size_t)(n0 + srow) * K + k0 + sseg * 16;
      async_ld16(lds_off_of(&Bs[buf][srow][sseg * 16]),     src);
      async_ld16(lds_off_of(&Bs[buf][srow][sseg * 16 + 8]), src + 8);
    }
  };
  auto stageA_flush = [&](int buf) {   // f32 path only: convert + ds_store
    if (A_F32) {
      v8bf lo, hi;
#pragma unroll
      for (int i = 0; i < 4; ++i) {
        lo[i]     = f32_to_bf16(regA[0][i]);
        lo[4 + i] = f32_to_bf16(regA[1][i]);
        hi[i]     = f32_to_bf16(regA[2][i]);
        hi[4 + i] = f32_to_bf16(regA[3][i]);
      }
      *(v8bf*)&As[buf][srow][sseg * 16]     = lo;
      *(v8bf*)&As[buf][srow][sseg * 16 + 8] = hi;
    }
  };
  auto stageB_flush = [&](int buf) {
    if (B_F32) {
      v8bf lo, hi;
#pragma unroll
      for (int i = 0; i < 4; ++i) {
        lo[i]     = f32_to_bf16(regB[0][i]);
        lo[4 + i] = f32_to_bf16(regB[1][i]);
        hi[i]     = f32_to_bf16(regB[2][i]);
        hi[4 + i] = f32_to_bf16(regB[3][i]);
      }
      *(v8bf*)&Bs[buf][srow][sseg * 16]     = lo;
      *(v8bf*)&Bs[buf][srow][sseg * 16 + 8] = hi;
    }
  };

  // ---------- pipelined main loop (1 barrier / K-step) ----------
  stageA_issue(0, 0);
  stageB_issue(0, 0);
  stageA_flush(0);
  stageB_flush(0);

  const int nsteps = K / BK;
  for (int i = 0; i < nsteps; ++i) {
    const int cur = i & 1;
    const int nxt = cur ^ 1;
    if (!A_F32 || !B_F32) wait_async0();   // my stage-i async copies done
    __syncthreads();                       // everyone's stage-i data visible

    const bool more = (i + 1 < nsteps);
    if (more) {                            // overlap next-stage movement with WMMA
      stageA_issue(nxt, (i + 1) * BK);
      stageB_issue(nxt, (i + 1) * BK);
    }

    // ---- WMMA: each wave does 4x2 fragments of 16x16, K=32 ----
    union Frag { v16bf v; v8bf h[2]; };
    Frag bfr[2];
#pragma unroll
    for (int ni = 0; ni < 2; ++ni) {
      const int n = wn * 32 + ni * 16 + lr;   // lane holds column n, K = half*8+0..7, 16+half*8+0..7
      bfr[ni].h[0] = *(const v8bf*)&Bs[cur][n][half * 8];
      bfr[ni].h[1] = *(const v8bf*)&Bs[cur][n][16 + half * 8];
    }
#pragma unroll
    for (int mi = 0; mi < 4; ++mi) {
      Frag af;
      const int r = wm * 64 + mi * 16 + lr;   // lane holds row r
      af.h[0] = *(const v8bf*)&As[cur][r][half * 8];
      af.h[1] = *(const v8bf*)&As[cur][r][16 + half * 8];
#pragma unroll
      for (int ni = 0; ni < 2; ++ni) {
        acc[mi][ni] = __builtin_amdgcn_wmma_f32_16x16x32_bf16(
            false, af.v, false, bfr[ni].v, (short)0, acc[mi][ni], false, false);
      }
    }

    if (more) {                            // f32 operands: convert + ds_store to other buffer
      stageA_flush(nxt);
      stageB_flush(nxt);
    }
  }

  // ---- epilogue: VGPR r, lanes 0-15 -> M=r, lanes 16-31 -> M=r+8; N=lr ----
#pragma unroll
  for (int mi = 0; mi < 4; ++mi) {
#pragma unroll
    for (int ni = 0; ni < 2; ++ni) {
#pragma unroll
      for (int r = 0; r < 8; ++r) {
        const int m = m0 + wm * 64 + mi * 16 + half * 8 + r;
        const int n = n0 + wn * 32 + ni * 16 + lr;
        float vv = acc[mi][ni][r] * scale;
        if (bias) vv += bias[n];
        if (OUT_MODE == 0) {
          ((float*)Cp)[(size_t)m * N + n] = vv;
        } else if (OUT_MODE == 1) {
          ((__bf16*)Cp)[(size_t)m * N + n] = f32_to_bf16(vv);
        } else {
          const int b = m / S, s = m % S;
          ((__bf16*)Cp)[(size_t)b * N * S + (size_t)n * S + s] = f32_to_bf16(vv);
        }
      }
    }
  }
}

__global__ __launch_bounds__(TPB)
void softmax_kernel(const float* __restrict__ sc, void* __restrict__ prp, int S)
{
  __bf16* pr = (__bf16*)prp;
  const int row = blockIdx.x;
  const float* r = sc + (size_t)row * S;
  __bf16* p = pr + (size_t)row * S;
  __shared__ float red[TPB];

  float mx = -3.402823466e38f;
  for (int c = threadIdx.x; c < S; c += TPB) mx = fmaxf(mx, r[c]);
  red[threadIdx.x] = mx;
  __syncthreads();
  for (int s = TPB / 2; s > 0; s >>= 1) {
    if (threadIdx.x < s) red[threadIdx.x] = fmaxf(red[threadIdx.x], red[threadIdx.x + s]);
    __syncthreads();
  }
  mx = red[0];
  __syncthreads();

  float sum = 0.f;
  for (int c = threadIdx.x; c < S; c += TPB) sum += __expf(r[c] - mx);
  red[threadIdx.x] = sum;
  __syncthreads();
  for (int s = TPB / 2; s > 0; s >>= 1) {
    if (threadIdx.x < s) red[threadIdx.x] += red[threadIdx.x + s];
    __syncthreads();
  }
  const float inv = 1.0f / red[0];
  for (int c = threadIdx.x; c < S; c += TPB)
    p[c] = f32_to_bf16(__expf(r[c] - mx) * inv);
}

extern "C" void kernel_launch(void* const* d_in, const int* in_sizes, int n_in,
                              void* d_out, int out_size, void* d_ws, size_t ws_size,
                              hipStream_t stream)
{
  (void)in_sizes; (void)n_in; (void)out_size; (void)ws_size;
  const float* k  = (const float*)d_in[0];
  const float* v  = (const float*)d_in[1];
  const float* q  = (const float*)d_in[2];
  const float* Wk = (const float*)d_in[3];
  const float* Wv = (const float*)d_in[4];
  const float* bv = (const float*)d_in[5];
  const float* Wq = (const float*)d_in[6];
  const float* bq = (const float*)d_in[7];
  const float* Wo = (const float*)d_in[8];
  const float* bo = (const float*)d_in[9];
  float* out = (float*)d_out;

  constexpr int    Bn = 4, S = 2048, D = 1024;
  constexpr size_t ME = (size_t)Bn * S * D;          // 8M elements
  char* ws = (char*)d_ws;
  void* kp   = (void*)(ws);                          // bf16 [B*S, D]      16 MB
  void* qp   = (void*)(ws + ME * 2);                 // bf16 [B*S, D]      16 MB
  void* vpT  = (void*)(ws + ME * 4);                 // bf16 [B][D][S]     16 MB
  void* ctx  = (void*)(ws + ME * 6);                 // bf16 [B*S, D]      16 MB
  float* scor = (float*)(ws + ME * 8);               // f32  [S, S]        16 MB
  void* prob = (void*)(ws + ME * 8 + (size_t)S * S * 4); // bf16 [S, S]     8 MB

  dim3 blk(TPB);
  dim3 gProj(D / BN, (Bn * S) / BM);                 // 8 x 64
  dim3 gScore(S / BN, S / BM);                       // 16 x 16
  dim3 gCtx(D / BN, S / BM);                         // 8 x 16

  // Projections: kp = k@Wk^T ; qp = q@Wq^T + bq ; vpT = (v@Wv^T + bv)^T per batch
  gemm_bt_kernel<1, 1, 1><<<gProj, blk, 0, stream>>>(k, Wk, kp, nullptr, 1.f, Bn * S, D, D, S);
  gemm_bt_kernel<1, 1, 1><<<gProj, blk, 0, stream>>>(q, Wq, qp, bq,      1.f, Bn * S, D, D, S);
  gemm_bt_kernel<1, 1, 2><<<gProj, blk, 0, stream>>>(v, Wv, vpT, bv,     1.f, Bn * S, D, D, S);

  const float scale = 0.03125f;                      // 1/sqrt(1024)
  for (int b = 0; b < Bn; ++b) {
    const char* qpb = (const char*)qp + (size_t)b * S * D * 2;
    const char* kpb = (const char*)kp + (size_t)b * S * D * 2;
    const char* vtb = (const char*)vpT + (size_t)b * D * S * 2;
    char*       ctb = (char*)ctx + (size_t)b * S * D * 2;
    // scores = scale * qp @ kp^T   (f32)
    gemm_bt_kernel<0, 0, 0><<<gScore, blk, 0, stream>>>(qpb, kpb, scor, nullptr, scale, S, S, D, S);
    // probs = softmax(scores)  -> bf16
    softmax_kernel<<<dim3(S), blk, 0, stream>>>(scor, prob, S);
    // ctx = probs @ vpT^T  (bf16)
    gemm_bt_kernel<0, 0, 1><<<gCtx, blk, 0, stream>>>(prob, vtb, ctb, nullptr, 1.f, S, D, S, S);
  }
  // out = ctx @ Wo^T + bo  (f32)
  gemm_bt_kernel<0, 1, 0><<<gProj, blk, 0, stream>>>(ctx, Wo, out, bo, 1.f, Bn * S, D, D, S);
}
